// SQVAE_18116172054713
// MI455X (gfx1250) — matrix operand order repaired and checked by reading, not compiled, not run on hardware
//
#include <hip/hip_runtime.h>
#include <hip/hip_bf16.h>

// SQVAE forward for MI455X (gfx1250, wave32, WMMA).
// GEMMs: 128x128x32 tiles, f16 WMMA (v_wmma_f32_16x16x32_f16) with f32 accum.
// Data movement: GLOBAL_LOAD_ASYNC_TO_LDS_B128 with manual double-buffering,
// gated by s_wait_asynccnt + workgroup barriers. All B operands pre-transposed
// to [Ncols x Kd] f16 so LDS staging is contiguous b128 for every GEMM.

typedef __attribute__((ext_vector_type(4)))  _Float16 v4h;
typedef __attribute__((ext_vector_type(8)))  _Float16 v8h;
typedef __attribute__((ext_vector_type(16))) _Float16 v16h;
typedef __attribute__((ext_vector_type(8)))  float    v8f;

#define BM 128
#define BN 128
#define BK 32

// ---------------- CDNA5 async global->LDS copy (16B per lane) ----------------
__device__ __forceinline__ void async_copy_b128(const void* gaddr, unsigned ldsoff) {
  asm volatile("global_load_async_to_lds_b128 %0, %1, off"
               :: "v"(ldsoff), "v"(gaddr) : "memory");
}
__device__ __forceinline__ void wait_async0() {
  asm volatile("s_wait_asynccnt 0x0" ::: "memory");
}
// low 32 bits of a generic shared pointer == LDS byte offset (ISA aperture rule)
__device__ __forceinline__ unsigned lds_off(const void* p) {
  return (unsigned)(unsigned long long)p;
}

// ---------------- block reduction helpers (blockDim.x == 256) ----------------
__device__ __forceinline__ float block_sum(float v, float* red) {
  const int tid = threadIdx.x;
  red[tid] = v; __syncthreads();
  #pragma unroll
  for (int s = 128; s > 0; s >>= 1) {
    if (tid < s) red[tid] += red[tid + s];
    __syncthreads();
  }
  float r = red[0]; __syncthreads();
  return r;
}

__device__ __forceinline__ float block_max(float v, float* red) {
  const int tid = threadIdx.x;
  red[tid] = v; __syncthreads();
  #pragma unroll
  for (int s = 128; s > 0; s >>= 1) {
    if (tid < s) red[tid] = fmaxf(red[tid], red[tid + s]);
    __syncthreads();
  }
  float r = red[0]; __syncthreads();
  return r;
}

// ---------------- WMMA GEMM: C = A(f16) @ Bt(f16)^T-layout, f32 accum --------
// A : [M x Kd] row-major f16.
// Bt: [Ncols x Kd] row-major f16 (i.e. logical B stored K-contiguous per col).
// mode 0: C = acc + bias[col] (bias nullable); optional f16 copy to Ch.
// mode 1 (logits): C = w * (2*acc - zn[row] - cn[col]), w = 0.5/clip(exp(lpq))
__global__ __launch_bounds__(256)
void wmma_gemm_kernel(const _Float16* __restrict__ A,
                      const _Float16* __restrict__ Bt,
                      const float* __restrict__ bias,
                      const float* __restrict__ zn,
                      const float* __restrict__ cn,
                      const float* __restrict__ lpq,
                      float* __restrict__ C,
                      _Float16* __restrict__ Ch,
                      int M, int Ncols, int Kd, int mode)
{
  __shared__ __align__(16) _Float16 sA[2][BM][BK];   // 2 x 8 KB
  __shared__ __align__(16) _Float16 sB[2][BN][BK];   // 2 x 8 KB

  const int tid  = threadIdx.x;
  const int wave = tid >> 5;
  const int lane = tid & 31;
  const int wm = wave >> 1;            // 0..3 : 32-row slab
  const int wn = wave & 1;             // 0..1 : 64-col slab
  const int hsel = lane >> 4;          // 0/1  : half-wave select
  const int l16  = lane & 15;
  const long blockRow = (long)blockIdx.y * BM;
  const long blockCol = (long)blockIdx.x * BN;

  // per-thread staging slots: 2 x 16B for A, 2 x 16B for B (4096 halves each)
  const int i0 = tid * 8,        r0s = i0 >> 5, c0s = i0 & 31;
  const int i1 = 2048 + tid * 8, r1s = i1 >> 5, c1s = i1 & 31;

  v8f acc[2][4];
  const v8f vzero = {0.f, 0.f, 0.f, 0.f, 0.f, 0.f, 0.f, 0.f};
  #pragma unroll
  for (int i = 0; i < 2; ++i)
    #pragma unroll
    for (int j = 0; j < 4; ++j)
      acc[i][j] = vzero;

  const int T = Kd / BK;

  // prologue: async-stage tile 0 into buffer 0
  {
    const int k0 = 0;
    async_copy_b128(A  + (blockRow + r0s) * (long)Kd + k0 + c0s, lds_off(&sA[0][r0s][c0s]));
    async_copy_b128(A  + (blockRow + r1s) * (long)Kd + k0 + c1s, lds_off(&sA[0][r1s][c1s]));
    async_copy_b128(Bt + (blockCol + r0s) * (long)Kd + k0 + c0s, lds_off(&sB[0][r0s][c0s]));
    async_copy_b128(Bt + (blockCol + r1s) * (long)Kd + k0 + c1s, lds_off(&sB[0][r1s][c1s]));
  }

  for (int t = 0; t < T; ++t) {
    wait_async0();          // tile t landed in LDS (this wave's async ops)
    __syncthreads();        // all waves' tile t visible; all reads of buf^1 done

    if (t + 1 < T) {        // overlap: async-stage tile t+1 into the other buffer
      const int k0 = (t + 1) * BK;
      const int nb = (t + 1) & 1;
      async_copy_b128(A  + (blockRow + r0s) * (long)Kd + k0 + c0s, lds_off(&sA[nb][r0s][c0s]));
      async_copy_b128(A  + (blockRow + r1s) * (long)Kd + k0 + c1s, lds_off(&sA[nb][r1s][c1s]));
      async_copy_b128(Bt + (blockCol + r0s) * (long)Kd + k0 + c0s, lds_off(&sB[nb][r0s][c0s]));
      async_copy_b128(Bt + (blockCol + r1s) * (long)Kd + k0 + c1s, lds_off(&sB[nb][r1s][c1s]));
    }

    const int buf = t & 1;
    // A 16x32 frag: lanes 0-15 hold K {0..7,16..23}; lanes 16-31 {8..15,24..31}
    v16h afrag[2];
    #pragma unroll
    for (int i = 0; i < 2; ++i) {
      const _Float16* base = &sA[buf][wm * 32 + i * 16 + l16][hsel * 8];
      v8h lo = *(const v8h*)base;
      v8h hi = *(const v8h*)(base + 16);
      afrag[i] = __builtin_shufflevector(lo, hi, 0,1,2,3,4,5,6,7,
                                                  8,9,10,11,12,13,14,15);
    }
    // B 32x16 frag: lanes 0-15 hold K {0..7,16..23}; lanes 16-31 {8..15,24..31} (col = lane&15)
    v16h bfrag[4];
    #pragma unroll
    for (int j = 0; j < 4; ++j) {
      const _Float16* base = &sB[buf][wn * 64 + j * 16 + l16][hsel * 8];
      v8h lo = *(const v8h*)base;
      v8h hi = *(const v8h*)(base + 16);
      bfrag[j] = __builtin_shufflevector(lo, hi, 0,1,2,3,4,5,6,7,
                                                  8,9,10,11,12,13,14,15);
    }
    #pragma unroll
    for (int i = 0; i < 2; ++i)
      #pragma unroll
      for (int j = 0; j < 4; ++j)
        acc[i][j] = __builtin_amdgcn_wmma_f32_16x16x32_f16(
            false, afrag[i], false, bfrag[j], (short)0, acc[i][j], false, false);

    __syncthreads();        // reads of buf done before it's overwritten at t+2
  }

  // ---- fused epilogue ----
  float w = 0.0f;
  if (mode == 1) {
    float vq = fmaxf(expf(lpq[0]), 1e-10f);
    w = 0.5f / vq;
  }
  #pragma unroll
  for (int i = 0; i < 2; ++i) {
    #pragma unroll
    for (int j = 0; j < 4; ++j) {
      #pragma unroll
      for (int r = 0; r < 8; ++r) {
        // C/D layout: VGPR r -> M = r + 8*(lane>=16); N = lane & 15
        const long row = blockRow + wm * 32 + i * 16 + hsel * 8 + r;
        const long col = blockCol + wn * 64 + j * 16 + l16;
        float v = acc[i][j][r];
        if (mode == 1)      v = w * (2.0f * v - zn[row] - cn[col]);
        else if (bias)      v += bias[col];
        C[row * (long)Ncols + col] = v;
        if (Ch) Ch[row * (long)Ncols + col] = (_Float16)v;
      }
    }
  }
}

// ---------------- transpose + f32->f16: out[C x R] = in[R x C]^T -------------
__global__ __launch_bounds__(256)
void transpose_f32_to_f16_kernel(const float* __restrict__ in,
                                 _Float16* __restrict__ out, int R, int C)
{
  __shared__ _Float16 tile[32][33];
  const int tx = threadIdx.x & 31, ty = threadIdx.x >> 5;   // 32 x 8
  const int c0 = blockIdx.x * 32, r0 = blockIdx.y * 32;
  #pragma unroll
  for (int i = 0; i < 32; i += 8)
    tile[ty + i][tx] = (_Float16)in[(size_t)(r0 + ty + i) * C + c0 + tx];
  __syncthreads();
  #pragma unroll
  for (int i = 0; i < 32; i += 8)
    out[(size_t)(c0 + ty + i) * R + r0 + tx] = tile[tx][ty + i];
}

// ---------------- fused ReLU + LayerNorm (+f16 recast, +row norm) ------------
// One block per row, D == 512 (2 elems/thread).
__global__ __launch_bounds__(256)
void relu_ln_kernel(const float* __restrict__ X,
                    const float* __restrict__ g, const float* __restrict__ be,
                    float* __restrict__ Y, _Float16* __restrict__ Yh,
                    float* __restrict__ rn, int D)
{
  __shared__ float red[256];
  const int tid = threadIdx.x;
  const size_t base = (size_t)blockIdx.x * D;
  float a = fmaxf(X[base + tid], 0.f);
  float b = fmaxf(X[base + tid + 256], 0.f);
  const float mean = block_sum(a + b, red) * (1.0f / 512.0f);
  const float var  = block_sum(a * a + b * b, red) * (1.0f / 512.0f) - mean * mean;
  const float inv  = rsqrtf(var + 1e-5f);
  const float y0 = (a - mean) * inv * g[tid] + be[tid];
  const float y1 = (b - mean) * inv * g[tid + 256] + be[tid + 256];
  if (Y)  { Y[base + tid] = y0; Y[base + tid + 256] = y1; }
  if (Yh) { Yh[base + tid] = (_Float16)y0; Yh[base + tid + 256] = (_Float16)y1; }
  if (rn) {
    const float s = block_sum(y0 * y0 + y1 * y1, red);
    if (tid == 0) rn[blockIdx.x] = s;
  }
}

// ---------------- codebook: f32->f16 + per-code squared norm -----------------
__global__ __launch_bounds__(256)
void codebook_kernel(const float* __restrict__ cb, _Float16* __restrict__ cbh,
                     float* __restrict__ cn, int D)
{
  __shared__ float red[256];
  const int k = blockIdx.x, tid = threadIdx.x;
  const size_t base = (size_t)k * D;
  const float a = cb[base + tid], b = cb[base + tid + 256];
  cbh[base + tid] = (_Float16)a;
  cbh[base + tid + 256] = (_Float16)b;
  const float s = block_sum(a * a + b * b, red);
  if (tid == 0) cn[k] = s;
}

// ---------------- fused log-softmax + gumbel-softmax over K = 1024 -----------
__global__ __launch_bounds__(256)
void softmax_gumbel_kernel(const float* __restrict__ logits,
                           const float* __restrict__ gu,
                           float* __restrict__ maxlse,
                           _Float16* __restrict__ enc,
                           float* __restrict__ kldd, int K)
{
  __shared__ float red[256];
  const int row = blockIdx.x, tid = threadIdx.x;
  const float* lr = logits + (size_t)row * K;
  const float* ur = gu + (size_t)row * K;
  float l[4], gl[4];
  float m1 = -3.0e38f, m2 = -3.0e38f;
  #pragma unroll
  for (int i = 0; i < 4; ++i) {
    l[i] = lr[tid + i * 256];
    const float u = ur[tid + i * 256];
    gl[i] = l[i] - logf(-logf(u + 1e-10f) + 1e-10f);   // (logit + gumbel)/TEMP, TEMP=1
    m1 = fmaxf(m1, l[i]);
    m2 = fmaxf(m2, gl[i]);
  }
  m1 = block_max(m1, red);
  m2 = block_max(m2, red);
  float s1 = 0.f, s2 = 0.f, e2[4];
  #pragma unroll
  for (int i = 0; i < 4; ++i) {
    s1 += expf(l[i] - m1);
    e2[i] = expf(gl[i] - m2);
    s2 += e2[i];
  }
  s1 = block_sum(s1, red);
  s2 = block_sum(s2, red);
  const float lse = logf(s1);
  const float inv2 = 1.0f / s2;
  float kd = 0.f;
  #pragma unroll
  for (int i = 0; i < 4; ++i) {
    const float lp = l[i] - m1 - lse;
    kd += expf(lp) * lp;
    enc[(size_t)row * K + tid + i * 256] = (_Float16)(e2[i] * inv2);
  }
  kd = block_sum(kd, red);
  if (tid == 0) {
    maxlse[2 * row] = m1;
    maxlse[2 * row + 1] = lse;
    atomicAdd(kldd, kd);
  }
}

// ---------------- avg_probs: column means of softmax(logits) -----------------
__global__ __launch_bounds__(256)
void avg_probs_kernel(const float* __restrict__ logits,
                      const float* __restrict__ maxlse,
                      float* __restrict__ avg, int K, float invN)
{
  const int tid = threadIdx.x;
  const int r0 = blockIdx.x * 32;
  float s0 = 0.f, s1 = 0.f, s2 = 0.f, s3 = 0.f;
  for (int r = 0; r < 32; ++r) {
    const int row = r0 + r;
    const float m = maxlse[2 * row], l = maxlse[2 * row + 1];
    const float* lr = logits + (size_t)row * K;
    s0 += expf(lr[tid      ] - m - l);
    s1 += expf(lr[tid + 256] - m - l);
    s2 += expf(lr[tid + 512] - m - l);
    s3 += expf(lr[tid + 768] - m - l);
  }
  atomicAdd(&avg[tid      ], s0 * invN);
  atomicAdd(&avg[tid + 256], s1 * invN);
  atomicAdd(&avg[tid + 512], s2 * invN);
  atomicAdd(&avg[tid + 768], s3 * invN);
}

// ---------------- sum (z - z_q)^2 --------------------------------------------
__global__ __launch_bounds__(256)
void kldc_kernel(const float* __restrict__ z, const float* __restrict__ zq,
                 float* __restrict__ out, size_t n)
{
  __shared__ float red[256];
  size_t i = (size_t)blockIdx.x * blockDim.x + threadIdx.x;
  const size_t stride = (size_t)gridDim.x * blockDim.x;
  float s = 0.f;
  for (; i < n; i += stride) {
    const float d = z[i] - zq[i];
    s += d * d;
  }
  s = block_sum(s, red);
  if (threadIdx.x == 0) atomicAdd(out, s);
}

// ---------------- scalars: loss + perplexity ---------------------------------
__global__ __launch_bounds__(256)
void finalize_kernel(const float* __restrict__ avg, const float* __restrict__ scal,
                     const float* __restrict__ lpq, float* __restrict__ out,
                     int K, float invB)
{
  __shared__ float red[256];
  const int tid = threadIdx.x;
  float s = 0.f;
  for (int i = tid; i < K; i += 256) {
    const float a = avg[i];
    s += a * logf(a + 1e-7f);
  }
  s = block_sum(s, red);
  if (tid == 0) {
    const float perp = expf(-s);
    const float vq = fmaxf(expf(lpq[0]), 1e-10f);
    const float w = 0.5f / vq;
    const float loss = scal[0] * invB + w * scal[1] * invB;
    out[0] = loss;
    out[1] = perp;
  }
}

// ---------------- misc -------------------------------------------------------
__global__ void f32_to_f16_kernel(const float* __restrict__ x,
                                  _Float16* __restrict__ y, size_t n)
{
  size_t i = (size_t)blockIdx.x * blockDim.x + threadIdx.x;
  const size_t st = (size_t)gridDim.x * blockDim.x;
  for (; i < n; i += st) y[i] = (_Float16)x[i];
}

__global__ void init_kernel(float* __restrict__ avg, float* __restrict__ scal, int K)
{
  const int i = blockIdx.x * blockDim.x + threadIdx.x;
  if (i < K) avg[i] = 0.f;
  if (i < 2) scal[i] = 0.f;
}

// =============================================================================
extern "C" void kernel_launch(void* const* d_in, const int* in_sizes, int n_in,
                              void* d_out, int out_size, void* d_ws, size_t ws_size,
                              hipStream_t stream)
{
  (void)in_sizes; (void)n_in; (void)out_size; (void)ws_size;
  const int N = 8 * 2048;   // 16384 rows
  const int D = 512;
  const int K = 1024;

  const float* x       = (const float*)d_in[0];
  const float* gu      = (const float*)d_in[1];
  const float* enc_w1  = (const float*)d_in[2];
  const float* enc_b1  = (const float*)d_in[3];
  const float* enc_g1  = (const float*)d_in[4];
  const float* enc_be1 = (const float*)d_in[5];
  const float* enc_w2  = (const float*)d_in[6];
  const float* enc_b2  = (const float*)d_in[7];
  const float* enc_g2  = (const float*)d_in[8];
  const float* enc_be2 = (const float*)d_in[9];
  const float* dec_w1  = (const float*)d_in[10];
  const float* dec_b1  = (const float*)d_in[11];
  const float* dec_g1  = (const float*)d_in[12];
  const float* dec_be1 = (const float*)d_in[13];
  const float* dec_w2  = (const float*)d_in[14];
  const float* dec_b2  = (const float*)d_in[15];
  const float* dec_g2  = (const float*)d_in[16];
  const float* dec_be2 = (const float*)d_in[17];
  const float* cb      = (const float*)d_in[18];
  const float* lpq     = (const float*)d_in[19];

  // ---- workspace carve-out ----
  char* ws = (char*)d_ws;
  size_t off = 0;
  auto alloc = [&](size_t bytes) -> void* {
    void* p = ws + off;
    off += (bytes + 255) & ~(size_t)255;
    return p;
  };
  _Float16* w1t  = (_Float16*)alloc((size_t)D * D * 2);   // enc_w1^T f16
  _Float16* w2t  = (_Float16*)alloc((size_t)D * D * 2);
  _Float16* dw1t = (_Float16*)alloc((size_t)D * D * 2);
  _Float16* dw2t = (_Float16*)alloc((size_t)D * D * 2);
  _Float16* cbh  = (_Float16*)alloc((size_t)K * D * 2);   // codebook f16 [K x D]
  _Float16* cbT  = (_Float16*)alloc((size_t)D * K * 2);   // codebook^T f16 [D x K]
  _Float16* xh   = (_Float16*)alloc((size_t)N * D * 2);   // x f16
  _Float16* ah0  = (_Float16*)alloc((size_t)N * D * 2);   // z f16
  _Float16* ah1  = (_Float16*)alloc((size_t)N * D * 2);   // hidden f16 ping-pong
  _Float16* zqh  = (_Float16*)alloc((size_t)N * D * 2);   // z_q f16
  _Float16* ench = (_Float16*)alloc((size_t)N * K * 2);   // gumbel encodings f16
  float* pre    = (float*)alloc((size_t)N * D * 4);       // GEMM pre-activation
  float* zf     = (float*)alloc((size_t)N * D * 4);       // z f32
  float* zqf    = (float*)alloc((size_t)N * D * 4);       // z_q f32
  float* logits = (float*)alloc((size_t)N * K * 4);
  float* mlse   = (float*)alloc((size_t)N * 2 * 4);
  float* znorm  = (float*)alloc((size_t)N * 4);
  float* cn     = (float*)alloc((size_t)K * 4);
  float* avg    = (float*)alloc((size_t)K * 4);
  float* scal   = (float*)alloc(2 * 4);                   // [kldd_sum, kldc_sum]

  const dim3 gD(D / BN, N / BM);   // (4,128)  512-col GEMMs
  const dim3 gK(K / BN, N / BM);   // (8,128)  logits GEMM
  const dim3 tDD(D / 32, D / 32);  // 512x512 weight transpose
  const dim3 tKD(D / 32, K / 32);  // codebook [K x D] -> [D x K]

  // ---- one-shot precision conversions / transposes + accumulator init ----
  init_kernel<<<(K + 255) / 256, 256, 0, stream>>>(avg, scal, K);
  transpose_f32_to_f16_kernel<<<tDD, 256, 0, stream>>>(enc_w1, w1t, D, D);
  transpose_f32_to_f16_kernel<<<tDD, 256, 0, stream>>>(enc_w2, w2t, D, D);
  transpose_f32_to_f16_kernel<<<tDD, 256, 0, stream>>>(dec_w1, dw1t, D, D);
  transpose_f32_to_f16_kernel<<<tDD, 256, 0, stream>>>(dec_w2, dw2t, D, D);
  transpose_f32_to_f16_kernel<<<tKD, 256, 0, stream>>>(cb, cbT, K, D);
  f32_to_f16_kernel<<<4096, 256, 0, stream>>>(x, xh, (size_t)N * D);
  codebook_kernel<<<K, 256, 0, stream>>>(cb, cbh, cn, D);

  // ---- encoder: LN(relu(x@W1+b1)) ; LN(relu(h@W2+b2)) = z ----
  wmma_gemm_kernel<<<gD, 256, 0, stream>>>(xh, w1t, enc_b1, nullptr, nullptr, nullptr,
                                           pre, nullptr, N, D, D, 0);
  relu_ln_kernel<<<N, 256, 0, stream>>>(pre, enc_g1, enc_be1, nullptr, ah1, nullptr, D);
  wmma_gemm_kernel<<<gD, 256, 0, stream>>>(ah1, w2t, enc_b2, nullptr, nullptr, nullptr,
                                           pre, nullptr, N, D, D, 0);
  relu_ln_kernel<<<N, 256, 0, stream>>>(pre, enc_g2, enc_be2, zf, ah0, znorm, D);

  // ---- logits = w*(2 z·c^T - ||z||^2 - ||c||^2); Bt = codebook [K x D] ----
  wmma_gemm_kernel<<<gK, 256, 0, stream>>>(ah0, cbh, nullptr, znorm, cn, lpq,
                                           logits, nullptr, N, K, D, 1);

  // ---- log-softmax + gumbel-softmax + discrete-KLD partials ----
  softmax_gumbel_kernel<<<N, 256, 0, stream>>>(logits, gu, mlse, ench, &scal[0], K);
  avg_probs_kernel<<<N / 32, 256, 0, stream>>>(logits, mlse, avg, K, 1.0f / (float)N);

  // ---- z_q = encodings @ codebook; Bt = codebook^T [D x K] ----
  wmma_gemm_kernel<<<gD, 256, 0, stream>>>(ench, cbT, nullptr, nullptr, nullptr, nullptr,
                                           zqf, zqh, N, D, K, 0);
  kldc_kernel<<<2048, 256, 0, stream>>>(zf, zqf, &scal[1], (size_t)N * D);

  // ---- decoder: LN(relu(z_q@V1+c1)) ; LN(relu(h@V2+c2)) = x_rec -> d_out ----
  wmma_gemm_kernel<<<gD, 256, 0, stream>>>(zqh, dw1t, dec_b1, nullptr, nullptr, nullptr,
                                           pre, nullptr, N, D, D, 0);
  relu_ln_kernel<<<N, 256, 0, stream>>>(pre, dec_g1, dec_be1, nullptr, ah1, nullptr, D);
  wmma_gemm_kernel<<<gD, 256, 0, stream>>>(ah1, dw2t, dec_b2, nullptr, nullptr, nullptr,
                                           pre, nullptr, N, D, D, 0);
  relu_ln_kernel<<<N, 256, 0, stream>>>(pre, dec_g2, dec_be2, (float*)d_out, nullptr,
                                        nullptr, D);

  // ---- loss + perplexity appended after x_rec ----
  finalize_kernel<<<1, 256, 0, stream>>>(avg, scal, lpq,
                                         (float*)d_out + (size_t)N * D, K, 0.125f);
}
